// FusionModule_5832565588677
// MI455X (gfx1250) — matrix operand order repaired and checked by reading, not compiled, not run on hardware
//
#include <hip/hip_runtime.h>
#include <hip/hip_bf16.h>
#include <math.h>

typedef __attribute__((ext_vector_type(16))) _Float16 v16h;
typedef __attribute__((ext_vector_type(8)))  _Float16 v8h;
typedef __attribute__((ext_vector_type(8)))  float    v8f;
typedef __attribute__((ext_vector_type(4)))  float    v4f;
typedef __attribute__((ext_vector_type(4)))  unsigned int u32x4;
typedef __attribute__((ext_vector_type(4)))  int      i32x4;
typedef __attribute__((ext_vector_type(8)))  int      i32x8;
typedef __attribute__((ext_vector_type(4)))  unsigned int uint4v;

union AF { v16h v; v8h h[2]; };   // A/B fragment, 16 f16 = 8 VGPRs
union CF { v8f  v; float f[8]; }; // C/D fragment, 8 f32

#define WMMA_F16(a, b, c) \
  __builtin_amdgcn_wmma_f32_16x16x32_f16(false, (a), false, (b), (short)0, (c), false, false)

static __device__ inline v8f zero8() {
  v8f z = {0.f, 0.f, 0.f, 0.f, 0.f, 0.f, 0.f, 0.f};
  return z;
}

#define NB    4
#define CIN   256
#define CO    128
#define NTOK  4096
#define QSCALE 0.08838834764831845f  // 1/sqrt(128)

#if __has_builtin(__builtin_amdgcn_tensor_load_to_lds)
#define HAVE_TDM 1
#else
#define HAVE_TDM 0
#endif

#if HAVE_TDM
// Build a D# and issue TENSOR_LOAD_TO_LDS for a 2D tile (dims in data_size
// units).  Bit packing per CDNA5 ISA §8.3 (group 0) / §8.4 (group 1).
static __device__ inline void tdm_load_2d(unsigned int lds_addr, const void* gptr,
                                          unsigned int ds_code,      // 0=1B,1=2B,2=4B,3=8B
                                          unsigned int pad_interval, // code: 0=2dw..7=256dw
                                          unsigned int pad_amount,   // code: n -> (n+1) dwords
                                          unsigned int tensor_d0, unsigned int tensor_d1,
                                          unsigned int tile_d0, unsigned int tile_d1,
                                          unsigned long long stride0) {
  const unsigned long long ga = (unsigned long long)(uintptr_t)gptr;
  u32x4 g0;
  g0.x = 1u;                                                  // count=1 user D#
  g0.y = lds_addr;                                            // LDS byte address
  g0.z = (unsigned int)(ga & 0xFFFFFFFFu);                    // global_addr[31:0]
  g0.w = (unsigned int)((ga >> 32) & 0x01FFFFFFu) | (2u << 30); // [56:32] | type=2
  i32x8 g1;
  g1[0] = (int)((ds_code << 16) | (1u << 20) |                // data_size | pad_enable
                (pad_interval << 22) | (pad_amount << 25));
  g1[1] = (int)((tensor_d0 & 0xFFFFu) << 16);                 // tensor_dim0[15:0]
  g1[2] = (int)(((tensor_d0 >> 16) & 0xFFFFu) | ((tensor_d1 & 0xFFFFu) << 16));
  g1[3] = (int)(((tensor_d1 >> 16) & 0xFFFFu) | ((tile_d0 & 0xFFFFu) << 16));
  g1[4] = (int)(tile_d1 & 0xFFFFu);                           // tile_dim1 (tile_dim2=0)
  g1[5] = (int)(unsigned int)(stride0 & 0xFFFFFFFFu);         // dim0_stride[31:0]
  g1[6] = (int)(unsigned int)((stride0 >> 32) & 0xFFFFu);     // dim0_stride[47:32]
  g1[7] = 0;
  i32x4 gz = {0, 0, 0, 0};
#if __has_include(<hip/amd_detail/amd_gfx1250_TDM.h>)
  i32x8 z8 = {0, 0, 0, 0, 0, 0, 0, 0};                        // 6-arg toolchain
  __builtin_amdgcn_tensor_load_to_lds(g0, g1, gz, gz, z8, 0);
#else
  __builtin_amdgcn_tensor_load_to_lds(g0, g1, gz, gz, 0);     // 5-arg toolchain
#endif
}

static __device__ inline void tdm_wait0() {
#if __has_builtin(__builtin_amdgcn_s_wait_tensorcnt)
  __builtin_amdgcn_s_wait_tensorcnt((short)0);
#else
  asm volatile("s_wait_tensorcnt 0x0" ::: "memory");
#endif
}
#endif  // HAVE_TDM

// ---------------------------------------------------------------------------
// Kernel 1: convert fp32 weights -> f16 once (weights stay L2-resident).
// ---------------------------------------------------------------------------
__global__ void cvt_w_kernel(const float* __restrict__ Wq, const float* __restrict__ Wk,
                             const float* __restrict__ Wv, const float* __restrict__ Wg,
                             _Float16* __restrict__ wq, _Float16* __restrict__ wk,
                             _Float16* __restrict__ wv, _Float16* __restrict__ wg) {
  int i = blockIdx.x * 256 + threadIdx.x;
  if (i < 32768) { wq[i] = (_Float16)Wq[i]; return; }
  i -= 32768;
  if (i < 32768) { wk[i] = (_Float16)Wk[i]; return; }
  i -= 32768;
  if (i < 32768) { wv[i] = (_Float16)Wv[i]; return; }
  i -= 32768;
  if (i < 16384) { wg[i] = (_Float16)Wg[i]; }
}

// ---------------------------------------------------------------------------
// Kernel 2: fused QKV projection.  One block = (batch b, 128 token rows).
// ---------------------------------------------------------------------------
__global__ __launch_bounds__(256)
void proj_qkv_kernel(const float* __restrict__ xp, const float* __restrict__ xf,
                     const _Float16* __restrict__ wq, const _Float16* __restrict__ wk,
                     const _Float16* __restrict__ wv,
                     const float* __restrict__ bq, const float* __restrict__ bk,
                     const float* __restrict__ bv,
                     _Float16* __restrict__ q16, _Float16* __restrict__ k16,
                     _Float16* __restrict__ v16t) {
  __shared__ __align__(16) _Float16 xs[128 * 40];  // [n][c], stride 40 (pad)

  const int b    = blockIdx.x >> 5;
  const int n0   = (blockIdx.x & 31) * 128;
  const int wave = threadIdx.x >> 5;
  const int lane = threadIdx.x & 31;
  const int col  = lane & 15;
  const int half = lane >> 4;

  CF aq[8], ak[8], av[8];
  #pragma unroll
  for (int t = 0; t < 8; ++t) { aq[t].v = zero8(); ak[t].v = zero8(); av[t].v = zero8(); }

  const int cl = threadIdx.x >> 3;        // local channel 0..31
  const int nb = (threadIdx.x & 7) * 16;  // n offset 0..112

  for (int kc = 0; kc < 8; ++kc) {        // 8 chunks of 32 channels (Cin=256)
    __syncthreads();
    const int cg = kc * 32 + cl;
    const float* src = (cg < 128)
        ? (xp + ((size_t)(b * 128 + cg)) * NTOK + n0 + nb)
        : (xf + ((size_t)(b * 128 + (cg - 128))) * NTOK + n0 + nb);
    #pragma unroll
    for (int i = 0; i < 16; i += 4) {
      v4f d = *(const v4f*)(src + i);
      xs[(nb + i + 0) * 40 + cl] = (_Float16)d.x;
      xs[(nb + i + 1) * 40 + cl] = (_Float16)d.y;
      xs[(nb + i + 2) * 40 + cl] = (_Float16)d.z;
      xs[(nb + i + 3) * 40 + cl] = (_Float16)d.w;
    }
    __syncthreads();

    AF a;
    const int row = wave * 16 + col;
    a.h[0] = *(const v8h*)&xs[row * 40 + half * 16];
    a.h[1] = *(const v8h*)&xs[row * 40 + half * 16 + 8];

    #pragma unroll
    for (int ct = 0; ct < 8; ++ct) {
      const int o  = ct * 16 + col;
      const int wo = o * CIN + kc * 32 + half * 16;
      v16h bqf = *(const v16h*)(wq + wo);
      aq[ct].v = WMMA_F16(a.v, bqf, aq[ct].v);
      v16h bkf = *(const v16h*)(wk + wo);
      ak[ct].v = WMMA_F16(a.v, bkf, ak[ct].v);
      v16h bvf = *(const v16h*)(wv + wo);
      av[ct].v = WMMA_F16(a.v, bvf, av[ct].v);
    }
  }

  #pragma unroll
  for (int ct = 0; ct < 8; ++ct) {
    const int o = ct * 16 + col;
    const float bqv = bq[o], bkv = bk[o], bvv = bv[o];
    #pragma unroll
    for (int r = 0; r < 8; ++r) {
      const int n = n0 + wave * 16 + r + 8 * half;
      const size_t qoff = ((size_t)(b * NTOK + n)) * CO + o;
      q16[qoff] = (_Float16)((aq[ct].f[r] + bqv) * QSCALE);
      k16[qoff] = (_Float16)(ak[ct].f[r] + bkv);
      v16t[((size_t)(b * CO + o)) * NTOK + n] = (_Float16)(av[ct].f[r] + bvv);
    }
  }
}

// ---------------------------------------------------------------------------
// Kernel 3: flash attention.  One block = (b, 128 query rows); wave owns 16
// rows.  K/V tiles for each 64-key chunk are DMA'd into LDS once per block by
// the Tensor Data Mover (TENSORcnt + workgroup barrier), then all 8 waves
// build B-fragments from LDS instead of 8x-duplicated global loads.
// K tile LDS pitch 272B (256B data + 16B TDM pad); V tile pitch 144B.
// ---------------------------------------------------------------------------
__global__ __launch_bounds__(256)
void attn_kernel(const _Float16* __restrict__ q16, const _Float16* __restrict__ k16,
                 const _Float16* __restrict__ v16t, _Float16* __restrict__ a16) {
  __shared__ __align__(16) _Float16 ks[64 * 136];     // 64 key rows x (128+8) f16
  __shared__ __align__(16) _Float16 vs[128 * 72];     // 128 o rows  x (64+8) f16
  __shared__ __align__(16) _Float16 pbuf[8][16][72];  // per-wave P relayout

  const int b    = blockIdx.x >> 5;
  const int n0   = (blockIdx.x & 31) * 128;
  const int wave = threadIdx.x >> 5;
  const int lane = threadIdx.x & 31;
  const int col  = lane & 15;
  const int half = lane >> 4;
  const int rowg = n0 + wave * 16 + col;

  AF qa[4];
  const _Float16* qrow = q16 + ((size_t)(b * NTOK + rowg)) * CO;
  #pragma unroll
  for (int kc = 0; kc < 4; ++kc) {
    qa[kc].h[0] = *(const v8h*)(qrow + kc * 32 + half * 16);
    qa[kc].h[1] = *(const v8h*)(qrow + kc * 32 + half * 16 + 8);
  }

  CF oacc[8];
  #pragma unroll
  for (int t = 0; t < 8; ++t) oacc[t].v = zero8();
  float mrow[8], lrow[8];
  #pragma unroll
  for (int r = 0; r < 8; ++r) { mrow[r] = -3.0e38f; lrow[r] = 0.f; }

#if HAVE_TDM
  const unsigned int ks_lds = (unsigned int)(uintptr_t)&ks[0];
  const unsigned int vs_lds = (unsigned int)(uintptr_t)&vs[0];
#endif

  for (int j0 = 0; j0 < NTOK; j0 += 64) {
    __syncthreads();  // all waves done with previous tile
#if HAVE_TDM
    if (wave == 0) {
      // K tile: 64 rows x 256B contiguous = 16KB, 8B elements, pad 16B/256B.
      tdm_load_2d(ks_lds, k16 + ((size_t)(b * NTOK + j0)) * CO,
                  /*ds=*/3u, /*pad_int=*/5u, /*pad_amt=*/3u,
                  /*t_d0=*/2048u, /*t_d1=*/1u, /*tile_d0=*/2048u, /*tile_d1=*/1u,
                  /*stride0=*/2048ull);
      // V tile: 128 rows x 128B, global row stride 8KB, 4B elems, pad 16B/128B.
      tdm_load_2d(vs_lds, v16t + ((size_t)b * CO) * NTOK + j0,
                  /*ds=*/2u, /*pad_int=*/4u, /*pad_amt=*/3u,
                  /*t_d0=*/32u, /*t_d1=*/128u, /*tile_d0=*/32u, /*tile_d1=*/128u,
                  /*stride0=*/2048ull);
      tdm_wait0();
    }
#else
    // Fallback: manual cooperative staging, 16B chunks, same padded layout.
    {
      const _Float16* kg = k16 + ((size_t)(b * NTOK + j0)) * CO;
      #pragma unroll
      for (int it = 0; it < 4; ++it) {
        const int c = threadIdx.x + it * 256;          // 1024 chunks of 16B
        const int j = c >> 4, off = (c & 15) * 8;
        *(uint4v*)&ks[j * 136 + off] = *(const uint4v*)(kg + j * CO + off);
      }
      const _Float16* vg = v16t + ((size_t)b * CO) * NTOK + j0;
      #pragma unroll
      for (int it = 0; it < 4; ++it) {
        const int c = threadIdx.x + it * 256;
        const int o = c >> 3, off = (c & 7) * 8;
        *(uint4v*)&vs[o * 72 + off] = *(const uint4v*)(vg + (size_t)o * NTOK + off);
      }
    }
#endif
    __syncthreads();  // tile published to all waves

    // S = Q * K^T for 16 rows x 64 keys (B-frags from LDS)
    CF st[4];
    #pragma unroll
    for (int jt = 0; jt < 4; ++jt) st[jt].v = zero8();
    #pragma unroll
    for (int jt = 0; jt < 4; ++jt) {
      const _Float16* krow = &ks[(jt * 16 + col) * 136];
      #pragma unroll
      for (int kc = 0; kc < 4; ++kc) {
        v16h kb = *(const v16h*)(krow + kc * 32 + half * 16);
        st[jt].v = WMMA_F16(qa[kc].v, kb, st[jt].v);
      }
    }

    // Online softmax: row m = r + 8*half lives across 16 lanes of a half
    float alpha[8];
    #pragma unroll
    for (int r = 0; r < 8; ++r) {
      float mx = fmaxf(fmaxf(st[0].f[r], st[1].f[r]), fmaxf(st[2].f[r], st[3].f[r]));
      #pragma unroll
      for (int sh = 8; sh >= 1; sh >>= 1) mx = fmaxf(mx, __shfl_xor(mx, sh, 32));
      const float mnew = fmaxf(mrow[r], mx);
      alpha[r] = __expf(mrow[r] - mnew);
      float psum = 0.f;
      #pragma unroll
      for (int jt = 0; jt < 4; ++jt) {
        const float p = __expf(st[jt].f[r] - mnew);
        st[jt].f[r] = p;
        psum += p;
      }
      #pragma unroll
      for (int sh = 8; sh >= 1; sh >>= 1) psum += __shfl_xor(psum, sh, 32);
      lrow[r] = lrow[r] * alpha[r] + psum;
      mrow[r] = mnew;
    }

    // Relayout P (C layout) -> A fragments through per-wave LDS scratch
    #pragma unroll
    for (int jt = 0; jt < 4; ++jt)
      #pragma unroll
      for (int r = 0; r < 8; ++r)
        pbuf[wave][r + 8 * half][jt * 16 + col] = (_Float16)st[jt].f[r];

    AF pa[2];
    #pragma unroll
    for (int s = 0; s < 2; ++s) {
      pa[s].h[0] = *(const v8h*)&pbuf[wave][col][s * 32 + half * 16];
      pa[s].h[1] = *(const v8h*)&pbuf[wave][col][s * 32 + half * 16 + 8];
    }

    // O = O*alpha + P @ V  (V B-frags from LDS)
    #pragma unroll
    for (int ct = 0; ct < 8; ++ct) {
      const _Float16* vrow = &vs[(ct * 16 + col) * 72];
      #pragma unroll
      for (int r = 0; r < 8; ++r) oacc[ct].f[r] *= alpha[r];
      #pragma unroll
      for (int s = 0; s < 2; ++s) {
        v16h vb = *(const v16h*)(vrow + s * 32 + half * 16);
        oacc[ct].v = WMMA_F16(pa[s].v, vb, oacc[ct].v);
      }
    }
  }

  // Normalize and store out_attn as f16 (B, N, Co)
  #pragma unroll
  for (int ct = 0; ct < 8; ++ct) {
    const int o = ct * 16 + col;
    #pragma unroll
    for (int r = 0; r < 8; ++r) {
      const int n = n0 + wave * 16 + r + 8 * half;
      a16[((size_t)(b * NTOK + n)) * CO + o] = (_Float16)(oacc[ct].f[r] / lrow[r]);
    }
  }
}

// ---------------------------------------------------------------------------
// Kernel 4: gate GEMM (Wg @ out_attn) + sigmoid + gated residual, fp32 out.
// ---------------------------------------------------------------------------
__global__ __launch_bounds__(256)
void gate_out_kernel(const _Float16* __restrict__ a16, const _Float16* __restrict__ wg,
                     const float* __restrict__ bg, const float* __restrict__ gammap,
                     const float* __restrict__ xp, float* __restrict__ out) {
  const int b    = blockIdx.x >> 5;
  const int n0   = (blockIdx.x & 31) * 128;
  const int wave = threadIdx.x >> 5;
  const int lane = threadIdx.x & 31;
  const int col  = lane & 15;
  const int half = lane >> 4;
  const int rowg = n0 + wave * 16 + col;
  const float gamma = gammap[0];

  AF aa[4];
  const _Float16* arow = a16 + ((size_t)(b * NTOK + rowg)) * CO;
  #pragma unroll
  for (int kc = 0; kc < 4; ++kc) {
    aa[kc].h[0] = *(const v8h*)(arow + kc * 32 + half * 16);
    aa[kc].h[1] = *(const v8h*)(arow + kc * 32 + half * 16 + 8);
  }

  CF g[8];
  #pragma unroll
  for (int t = 0; t < 8; ++t) g[t].v = zero8();
  #pragma unroll
  for (int ct = 0; ct < 8; ++ct) {
    const int o = ct * 16 + col;
    #pragma unroll
    for (int kc = 0; kc < 4; ++kc) {
      v16h wb = *(const v16h*)(wg + o * CO + kc * 32 + half * 16);
      g[ct].v = WMMA_F16(aa[kc].v, wb, g[ct].v);
    }
  }

  #pragma unroll
  for (int ct = 0; ct < 8; ++ct) {
    const int o = ct * 16 + col;
    const float bgo = bg[o];
    #pragma unroll
    for (int r = 0; r < 8; ++r) {
      const int n  = n0 + wave * 16 + r + 8 * half;
      const float av   = (float)a16[((size_t)(b * NTOK + n)) * CO + o];
      const float gate = 1.f / (1.f + __expf(-(g[ct].f[r] + bgo)));
      const float res  = xp[((size_t)(b * CO + o)) * NTOK + n];
      out[((size_t)(b * CO + o)) * NTOK + n] = gamma * gate * av + (1.f - gate) * res;
    }
  }
}

// ---------------------------------------------------------------------------
extern "C" void kernel_launch(void* const* d_in, const int* in_sizes, int n_in,
                              void* d_out, int out_size, void* d_ws, size_t ws_size,
                              hipStream_t stream) {
  (void)in_sizes; (void)n_in; (void)out_size; (void)ws_size;

  const float* xp    = (const float*)d_in[0];
  const float* xf    = (const float*)d_in[1];
  const float* Wq    = (const float*)d_in[2];
  const float* bq    = (const float*)d_in[3];
  const float* Wk    = (const float*)d_in[4];
  const float* bk    = (const float*)d_in[5];
  const float* Wv    = (const float*)d_in[6];
  const float* bv    = (const float*)d_in[7];
  const float* Wg    = (const float*)d_in[8];
  const float* bg    = (const float*)d_in[9];
  const float* gamma = (const float*)d_in[10];
  float* out = (float*)d_out;

  char* ws = (char*)d_ws;
  _Float16* wq16 = (_Float16*)(ws + 0);
  _Float16* wk16 = (_Float16*)(ws + 65536);
  _Float16* wv16 = (_Float16*)(ws + 131072);
  _Float16* wg16 = (_Float16*)(ws + 196608);
  _Float16* q16  = (_Float16*)(ws + 229376);
  _Float16* k16  = (_Float16*)(ws + 229376 + 1 * 4194304);
  _Float16* v16t = (_Float16*)(ws + 229376 + 2 * 4194304);
  _Float16* a16  = (_Float16*)(ws + 229376 + 3 * 4194304);

  cvt_w_kernel<<<448, 256, 0, stream>>>(Wq, Wk, Wv, Wg, wq16, wk16, wv16, wg16);
  proj_qkv_kernel<<<128, 256, 0, stream>>>(xp, xf, wq16, wk16, wv16, bq, bk, bv,
                                           q16, k16, v16t);
  attn_kernel<<<128, 256, 0, stream>>>(q16, k16, v16t, a16);
  gate_out_kernel<<<128, 256, 0, stream>>>(a16, wg16, bg, gamma, xp, out);
}
